// VllmMixtureOfExpertsOp_627065225257
// MI455X (gfx1250) — matrix operand order repaired and checked by reading, not compiled, not run on hardware
//
#include <hip/hip_runtime.h>
#include <hip/hip_bf16.h>
#include <math.h>

typedef __bf16 bf16;
typedef __attribute__((ext_vector_type(8)))  __bf16 v8bf;
typedef __attribute__((ext_vector_type(16))) __bf16 v16bf;
typedef __attribute__((ext_vector_type(8)))  float  v8f;

#define TT   8192
#define HH   2048
#define II   4096
#define EE   8
#define TOPK 2
#define NENT (TT * TOPK)          // 16384 token-expert entries total
#define MT   128                  // macro tile rows (entries)
#define NT   64                   // macro tile cols (output features)
#define KT   32                   // K step (matches wmma 16x16x32)
#define LDA  56                   // padded LDS row pitch in bf16 (112B, 16B-aligned, conflict-free)
#define MAX_MT (NENT / MT)        // 128 worst-case m-tiles per expert

__device__ __forceinline__ v8f wmma_bf16(v16bf a, v16bf b, v8f c) {
  // D = A(16x32) * B(32x16) + C, f32 accumulate
  return __builtin_amdgcn_wmma_f32_16x16x32_bf16(false, a, false, b, (short)0, c,
                                                 false, false);
}

// A operand (16x32 bf16): lane m=lane&15 holds row m; lanes0-15: K {0-7,16-23},
// lanes16-31: K {8-15,24-31}  (ISA 7.12.2)
__device__ __forceinline__ v16bf frag_a(const bf16* s, int row0, int lane) {
  const int m = lane & 15, h = lane >> 4;
  const bf16* p = s + (row0 + m) * LDA;
  v8bf lo = *(const v8bf*)(p + h * 8);
  v8bf hi = *(const v8bf*)(p + 16 + h * 8);
  return __builtin_shufflevector(lo, hi, 0,1,2,3,4,5,6,7,8,9,10,11,12,13,14,15);
}

// B operand (32x16 bf16): lane n=lane&15 holds column n; lanes0-15: K {0-7,16-23},
// lanes16-31: K {8-15,24-31} (same as A).  Tile stored row-major [N][K].
__device__ __forceinline__ v16bf frag_b(const bf16* s, int row0, int lane) {
  const int n = lane & 15, h = lane >> 4;
  const bf16* p = s + (row0 + n) * LDA + h * 8;
  v8bf lo = *(const v8bf*)(p);
  v8bf hi = *(const v8bf*)(p + 16);
  return __builtin_shufflevector(lo, hi, 0,1,2,3,4,5,6,7,8,9,10,11,12,13,14,15);
}

__device__ __forceinline__ void cvt_store4(bf16* d, float4 v) {
  d[0] = (bf16)v.x; d[1] = (bf16)v.y; d[2] = (bf16)v.z; d[3] = (bf16)v.w;
}

// ---------------- routing ----------------
// meta[0..7]=counts  meta[8..15]=offsets  meta[16..23]=cursor
__global__ void moe_zero_meta(int* meta) {
  if (threadIdx.x < 24) meta[threadIdx.x] = 0;
}

__global__ void moe_count(const int* __restrict__ route, int* meta) {
  const int t = blockIdx.x * 256 + threadIdx.x;
  if (t >= TT) return;
  atomicAdd(&meta[route[t * TOPK + 0]], 1);
  atomicAdd(&meta[route[t * TOPK + 1]], 1);
}

__global__ void moe_scan(int* meta) {
  if (threadIdx.x == 0) {
    int off = 0;
    for (int e = 0; e < EE; ++e) {
      meta[8 + e]  = off;
      meta[16 + e] = off;
      off += meta[e];
    }
  }
}

__global__ void moe_fill(const int* __restrict__ route, const float* __restrict__ rw,
                         int* meta, int* __restrict__ etok,
                         float* __restrict__ ecoef, int* __restrict__ edst) {
  const int t = blockIdx.x * 256 + threadIdx.x;
  if (t >= TT) return;
  for (int k = 0; k < TOPK; ++k) {
    const int e = route[t * TOPK + k];
    const int p = atomicAdd(&meta[16 + e], 1);
    etok[p]  = t;
    ecoef[p] = rw[t * TOPK + k];
    edst[p]  = t * TOPK + k;
  }
}

// ---------------- GEMM1: gate/up + SwiGLU, write bf16 h ----------------
__global__ __launch_bounds__(256)
void moe_gemm1(const float* __restrict__ x, const float* __restrict__ w13,
               const int* __restrict__ etok, const float* __restrict__ ecoef,
               const int* __restrict__ meta, bf16* __restrict__ hbuf) {
  const int e  = blockIdx.x / MAX_MT;
  const int mt = blockIdx.x % MAX_MT;
  const int Ne = meta[e];
  const int mb = mt * MT;
  if (mb >= Ne) return;
  const int eb = meta[8 + e];
  const int n0 = blockIdx.y * NT;

  __shared__ __align__(16) bf16 sA [2][MT * LDA];
  __shared__ __align__(16) bf16 sBg[2][NT * LDA];
  __shared__ __align__(16) bf16 sBu[2][NT * LDA];
  __shared__ int   sTok[MT];
  __shared__ float sCf [MT];

  const int tid = threadIdx.x;
  for (int r = tid; r < MT; r += 256) {
    const int row = mb + r;
    int t = 0; float c = 0.f;
    if (row < Ne) { t = etok[eb + row]; c = ecoef[eb + row]; }
    sTok[r] = t; sCf[r] = c;
  }
  __syncthreads();

  const float* wg = w13 + (size_t)e * (2 * II) * HH + (size_t)n0 * HH;
  const float* wu = wg + (size_t)II * HH;

  // Hoist all loop-invariant gather/stage addresses out of the K loop.
  const float* aSrc[4]; int aOff[4];
#pragma unroll
  for (int i = 0; i < 4; ++i) {
    const int c = tid + i * 256;
    const int row = c >> 3, f4 = c & 7;         // 128 rows x 8 float4 chunks
    aSrc[i] = x + (size_t)sTok[row] * HH + f4 * 4;
    aOff[i] = row * LDA + f4 * 4;
  }
  const float* gSrc[2]; const float* uSrc[2]; int bOff[2];
#pragma unroll
  for (int i = 0; i < 2; ++i) {
    const int c = tid + i * 256;
    const int row = c >> 3, f4 = c & 7;         // 64 rows x 8 float4 chunks
    gSrc[i] = wg + (size_t)row * HH + f4 * 4;
    uSrc[i] = wu + (size_t)row * HH + f4 * 4;
    bOff[i] = row * LDA + f4 * 4;
  }

  auto stage = [&](int buf, int kb) {
#pragma unroll
    for (int i = 0; i < 4; ++i)
      cvt_store4(&sA[buf][aOff[i]], *(const float4*)(aSrc[i] + kb));
#pragma unroll
    for (int i = 0; i < 2; ++i) {
      const float4 vg = *(const float4*)(gSrc[i] + kb);
      const float4 vu = *(const float4*)(uSrc[i] + kb);
      if (kb + KT < HH) {                 // global_prefetch_b8 of slab after next
        __builtin_prefetch(gSrc[i] + kb + KT, 0, 1);
        __builtin_prefetch(uSrc[i] + kb + KT, 0, 1);
      }
      cvt_store4(&sBg[buf][bOff[i]], vg);
      cvt_store4(&sBu[buf][bOff[i]], vu);
    }
  };

  const int lane = tid & 31, wave = tid >> 5;
  const int wm = wave >> 1, wn = wave & 1;    // 4x2 wave grid over 128x64 tile

  const v8f z = {0.f, 0.f, 0.f, 0.f, 0.f, 0.f, 0.f, 0.f};
  v8f ag[2][2], au[2][2];
  ag[0][0] = z; ag[0][1] = z; ag[1][0] = z; ag[1][1] = z;
  au[0][0] = z; au[0][1] = z; au[1][0] = z; au[1][1] = z;

  stage(0, 0);
  __syncthreads();

  int cur = 0;
  for (int kb = 0; kb < HH; kb += KT) {
    if (kb + KT < HH) stage(cur ^ 1, kb + KT);   // fill other buffer while computing

    const v16bf a0 = frag_a(sA[cur], wm * 32, lane);
    const v16bf a1 = frag_a(sA[cur], wm * 32 + 16, lane);
    const v16bf g0 = frag_b(sBg[cur], wn * 32, lane);
    const v16bf g1 = frag_b(sBg[cur], wn * 32 + 16, lane);
    const v16bf u0 = frag_b(sBu[cur], wn * 32, lane);
    const v16bf u1 = frag_b(sBu[cur], wn * 32 + 16, lane);

    ag[0][0] = wmma_bf16(a0, g0, ag[0][0]);
    ag[0][1] = wmma_bf16(a0, g1, ag[0][1]);
    ag[1][0] = wmma_bf16(a1, g0, ag[1][0]);
    ag[1][1] = wmma_bf16(a1, g1, ag[1][1]);
    au[0][0] = wmma_bf16(a0, u0, au[0][0]);
    au[0][1] = wmma_bf16(a0, u1, au[0][1]);
    au[1][0] = wmma_bf16(a1, u0, au[1][0]);
    au[1][1] = wmma_bf16(a1, u1, au[1][1]);

    __syncthreads();   // next-buffer stores done; cur reads done -> safe to swap
    cur ^= 1;
  }

  // epilogue: h = silu(g)*u * coef  (coef folded here; down-proj is linear)
  const int moff = (lane >> 4) * 8;
  const int ncol = lane & 15;
#pragma unroll
  for (int i = 0; i < 2; ++i) {
#pragma unroll
    for (int j = 0; j < 2; ++j) {
      const v8f g = ag[i][j], u = au[i][j];
#pragma unroll
      for (int r = 0; r < 8; ++r) {
        const int rloc = wm * 32 + i * 16 + moff + r;
        const int grow = mb + rloc;
        if (grow < Ne) {
          const float gv = g[r];
          const float s  = gv / (1.f + __expf(-gv));
          const float hv = s * u[r] * sCf[rloc];
          const int col  = n0 + wn * 32 + j * 16 + ncol;
          hbuf[(size_t)(eb + grow) * II + col] = (bf16)hv;
        }
      }
    }
  }
}

// ---------------- GEMM2: h @ w2^T, scatter f32 per-entry rows ----------------
__global__ __launch_bounds__(256)
void moe_gemm2(const bf16* __restrict__ hbuf, const float* __restrict__ w2,
               const int* __restrict__ edst, const int* __restrict__ meta,
               float* __restrict__ oscr) {
  const int e  = blockIdx.x / MAX_MT;
  const int mt = blockIdx.x % MAX_MT;
  const int Ne = meta[e];
  const int mb = mt * MT;
  if (mb >= Ne) return;
  const int eb = meta[8 + e];
  const int n0 = blockIdx.y * NT;

  __shared__ __align__(16) bf16 sA[2][MT * LDA];
  __shared__ __align__(16) bf16 sB[2][NT * LDA];
  __shared__ int sDst[MT];

  const int tid = threadIdx.x;
  for (int r = tid; r < MT; r += 256) {
    const int row = mb + r;
    sDst[r] = (row < Ne) ? edst[eb + row] : 0;
  }

  const float* wv = w2 + (size_t)e * HH * II + (size_t)n0 * II;

  // Loop-invariant addresses.
  const bf16* aSrc[2]; int aOff[2];
#pragma unroll
  for (int i = 0; i < 2; ++i) {
    const int c = tid + i * 256;
    const int row = c >> 2, k8 = c & 3;         // 128 rows x 4 v8bf chunks
    int rr = mb + row; if (rr >= Ne) rr = Ne - 1;   // clamp: stay in-bounds
    aSrc[i] = hbuf + (size_t)(eb + rr) * II + k8 * 8;
    aOff[i] = row * LDA + k8 * 8;
  }
  const float* bSrc[2]; int bOff[2];
#pragma unroll
  for (int i = 0; i < 2; ++i) {
    const int c = tid + i * 256;
    const int row = c >> 3, f4 = c & 7;         // 64 rows x 8 float4 chunks
    bSrc[i] = wv + (size_t)row * II + f4 * 4;
    bOff[i] = row * LDA + f4 * 4;
  }

  auto stage = [&](int buf, int kb) {
#pragma unroll
    for (int i = 0; i < 2; ++i)
      *(v8bf*)&sA[buf][aOff[i]] = *(const v8bf*)(aSrc[i] + kb);
#pragma unroll
    for (int i = 0; i < 2; ++i) {
      const float4 v = *(const float4*)(bSrc[i] + kb);
      if (kb + KT < II) __builtin_prefetch(bSrc[i] + kb + KT, 0, 1);
      cvt_store4(&sB[buf][bOff[i]], v);
    }
  };

  const int lane = tid & 31, wave = tid >> 5;
  const int wm = wave >> 1, wn = wave & 1;

  const v8f z = {0.f, 0.f, 0.f, 0.f, 0.f, 0.f, 0.f, 0.f};
  v8f acc[2][2];
  acc[0][0] = z; acc[0][1] = z; acc[1][0] = z; acc[1][1] = z;

  stage(0, 0);
  __syncthreads();

  int cur = 0;
  for (int kb = 0; kb < II; kb += KT) {
    if (kb + KT < II) stage(cur ^ 1, kb + KT);

    const v16bf a0 = frag_a(sA[cur], wm * 32, lane);
    const v16bf a1 = frag_a(sA[cur], wm * 32 + 16, lane);
    const v16bf b0 = frag_b(sB[cur], wn * 32, lane);
    const v16bf b1 = frag_b(sB[cur], wn * 32 + 16, lane);

    acc[0][0] = wmma_bf16(a0, b0, acc[0][0]);
    acc[0][1] = wmma_bf16(a0, b1, acc[0][1]);
    acc[1][0] = wmma_bf16(a1, b0, acc[1][0]);
    acc[1][1] = wmma_bf16(a1, b1, acc[1][1]);

    __syncthreads();
    cur ^= 1;
  }

  const int moff = (lane >> 4) * 8;
  const int ncol = lane & 15;
#pragma unroll
  for (int i = 0; i < 2; ++i) {
#pragma unroll
    for (int j = 0; j < 2; ++j) {
      const v8f a = acc[i][j];
#pragma unroll
      for (int r = 0; r < 8; ++r) {
        const int rloc = wm * 32 + i * 16 + moff + r;
        const int grow = mb + rloc;
        if (grow < Ne) {
          const int col = n0 + wn * 32 + j * 16 + ncol;
          oscr[(size_t)sDst[rloc] * HH + col] = a[r];
        }
      }
    }
  }
}

// ---------------- combine: out[t] = slot(t,0) + slot(t,1) ----------------
__global__ __launch_bounds__(256)
void moe_combine(const float* __restrict__ oscr, float* __restrict__ out) {
  const size_t i = (size_t)blockIdx.x * 256 + threadIdx.x;   // float4 index
  if (i >= (size_t)TT * HH / 4) return;
  const size_t t = (i * 4) / HH;
  const size_t h = (i * 4) % HH;
  const float4 a = *(const float4*)(oscr + (2 * t) * HH + h);
  const float4 b = *(const float4*)(oscr + (2 * t + 1) * HH + h);
  float4 r;
  r.x = a.x + b.x; r.y = a.y + b.y; r.z = a.z + b.z; r.w = a.w + b.w;
  *(float4*)(out + i * 4) = r;
}

extern "C" void kernel_launch(void* const* d_in, const int* in_sizes, int n_in,
                              void* d_out, int out_size, void* d_ws, size_t ws_size,
                              hipStream_t stream) {
  (void)in_sizes; (void)n_in; (void)out_size; (void)ws_size;
  const float* x     = (const float*)d_in[0];
  const int*   route = (const int*)d_in[1];
  const float* rw    = (const float*)d_in[2];
  const float* w13   = (const float*)d_in[3];
  const float* w2    = (const float*)d_in[4];
  float* out = (float*)d_out;

  char* ws = (char*)d_ws;
  int*   meta  = (int*)ws;                                   // 24 ints (256B reserved)
  int*   etok  = (int*)(ws + 256);                           // 64KB
  float* ecoef = (float*)(ws + 256 + NENT * 4);              // 64KB
  int*   edst  = (int*)(ws + 256 + NENT * 8);                // 64KB
  bf16*  hbuf  = (bf16*)(ws + 256 + NENT * 12);              // 128MB bf16 h
  float* oscr  = (float*)(ws + 256 + NENT * 12 + (size_t)NENT * II * 2);  // 128MB f32

  moe_zero_meta<<<1, 32, 0, stream>>>(meta);
  moe_count<<<TT / 256, 256, 0, stream>>>(route, meta);
  moe_scan<<<1, 32, 0, stream>>>(meta);
  moe_fill<<<TT / 256, 256, 0, stream>>>(route, rw, meta, etok, ecoef, edst);

  dim3 g1(EE * MAX_MT, II / NT);
  moe_gemm1<<<g1, 256, 0, stream>>>(x, w13, etok, ecoef, meta, hbuf);

  dim3 g2(EE * MAX_MT, HH / NT);
  moe_gemm2<<<g2, 256, 0, stream>>>(hbuf, w2, edst, meta, oscr);

  moe_combine<<<(TT * HH / 4 + 255) / 256, 256, 0, stream>>>(oscr, out);
}
